// RNN_43258910605702
// MI455X (gfx1250) — matrix operand order repaired and checked by reading, not compiled, not run on hardware
//
#include <hip/hip_runtime.h>
#include <hip/hip_bf16.h>
#include <math.h>

// ---------------------------------------------------------------------------
// Problem constants (from reference)
// ---------------------------------------------------------------------------
#define DD      300
#define DPAD    320      // K padding for X / gat W (10 k-tiles of 32)
#define HCPAD   304      // N padding for h (19 n-tiles of 16)
#define NAREA   32
#define EE      64
#define BB      32
#define TT      64
#define HH      4
#define CC      75
#define GG      (BB*TT)  // 2048
#define SIGPAD  608      // 600 -> 19 k-tiles of 32
#define NGLOB   50000
#define NT2     3125     // 50000/16 n-tiles in classifier

typedef __attribute__((ext_vector_type(16))) __bf16 v16bf;
typedef __attribute__((ext_vector_type(8)))  __bf16 v8bf;
typedef __attribute__((ext_vector_type(8)))  float  v8f;

// ---------------------------------------------------------------------------
// WMMA fragment helpers (CDNA5 ISA 7.12.2 layouts, wave32)
// A 16x32 bf16: lane<16 -> M=lane, K={0..7,16..23}; lane>=16 -> K={8..15,24..31}
//   => per lane: two aligned 16B loads.
// B 32x16 bf16: pre-packed fragment-major, 32 lanes x 16 contiguous bf16
//   => per lane: one 32B contiguous read (2 x b128), wave reads 1KB contiguous.
// C 16x16 f32: elem r -> (M = r + 8*(lane>=16), N = lane&15)
// ---------------------------------------------------------------------------
__device__ __forceinline__ v16bf ldA(const __bf16* base, int stride) {
  int lane = threadIdx.x & 31;
  const __bf16* row = base + (lane & 15) * stride + (lane >> 4) * 8;
  v8bf lo = *(const v8bf*)row;
  v8bf hi = *(const v8bf*)(row + 16);
  return __builtin_shufflevector(lo, hi, 0,1,2,3,4,5,6,7,8,9,10,11,12,13,14,15);
}

__device__ __forceinline__ v16bf ldBp(const __bf16* tile) {
  int lane = threadIdx.x & 31;
  const v8bf* p = (const v8bf*)(tile + lane * 16);
  v8bf lo = p[0], hi = p[1];
  return __builtin_shufflevector(lo, hi, 0,1,2,3,4,5,6,7,8,9,10,11,12,13,14,15);
}

__device__ __forceinline__ v8f wmma_bf(v16bf a, v16bf b, v8f c) {
  return __builtin_amdgcn_wmma_f32_16x16x32_bf16(false, a, false, b,
                                                 (short)0, c, false, false);
}

// ---------------------------------------------------------------------------
// Packing kernels
// ---------------------------------------------------------------------------
__global__ void k_cvtX(const float* __restrict__ X, __bf16* __restrict__ Xbf) {
  int row = blockIdx.x;
  int k = threadIdx.x;                 // 320 threads
  float v = (k < DD) ? X[(long)row * DD + k] : 0.f;
  Xbf[(long)row * DPAD + k] = (__bf16)v;
}

// Pack a weight matrix into WMMA-B fragment-major tiles.
// Logical B is [K x N]; source is either W[k*ldW + n] (trans=0) or
// W[n*ldW + k] (trans=1, torch [N][K] weights).
// out tile (kt,nt): 512 bf16; element (lane,e): k=kt*32+(lane>>4)*8+(e&7)+(e>>3)*16,
//                                               n=nt*16+(lane&15)
__global__ void k_packFrag(const float* __restrict__ W, __bf16* __restrict__ out,
                           int K, int N, int KT, int NT, int ldW, int trans) {
  long i = (long)blockIdx.x * blockDim.x + threadIdx.x;
  long total = (long)KT * NT * 512;
  if (i >= total) return;
  int tile = (int)(i >> 9);
  int rem  = (int)(i & 511);
  int lane = rem >> 4, e = rem & 15;
  int kt = tile / NT, nt = tile % NT;
  int k = kt * 32 + (lane >> 4) * 8 + (e & 7) + (e >> 3) * 16;
  int n = nt * 16 + (lane & 15);
  float v = 0.f;
  if (k < K && n < N) v = trans ? W[(long)n * ldW + k] : W[(long)k * ldW + n];
  out[i] = (__bf16)v;
}

// ---------------------------------------------------------------------------
// GAT linear: per graph, h[32 x 300] = gather(X)[32 x 300] @ W[300 x 300]
// ---------------------------------------------------------------------------
__global__ void k_gatGemm(const int* __restrict__ bxi, const __bf16* __restrict__ Xbf,
                          const __bf16* __restrict__ Wgp, __bf16* __restrict__ hbf) {
  __shared__ __align__(16) __bf16 Alds[NAREA * DPAD];    // 20 KB
  int g = blockIdx.x;
  int tid = threadIdx.x;
  // vectorized gather: 32 rows x 40 v8bf each
  for (int e = tid; e < NAREA * (DPAD / 8); e += 256) {
    int node = e / (DPAD / 8), kv = e % (DPAD / 8);
    int row = bxi[g * NAREA + node];
    ((v8bf*)Alds)[e] = ((const v8bf*)(Xbf + (long)row * DPAD))[kv];
  }
  __syncthreads();
  int w = tid >> 5;
  __bf16* outb = hbf + (long)g * NAREA * HCPAD;
  for (int tile = w; tile < 38; tile += 8) {   // 2 mt x 19 nt
    int mt = tile & 1, nt = tile >> 1;
    v8f acc = {};
#pragma unroll
    for (int kt = 0; kt < DPAD / 32; ++kt) {
      v16bf a = ldA(Alds + mt * 16 * DPAD + kt * 32, DPAD);
      v16bf b = ldBp(Wgp + (((long)kt * 19 + nt) << 9));
      acc = wmma_bf(a, b, acc);
    }
    int lane = tid & 31, n = lane & 15, mh = (lane >> 4) * 8;
#pragma unroll
    for (int r = 0; r < 8; ++r) {
      int m = mt * 16 + mh + r;
      outb[m * HCPAD + nt * 16 + n] = (__bf16)acc[r];
    }
  }
}

// ---------------------------------------------------------------------------
// Attention for node 0 + concat(word_emb, node0) -> signals [T][B][608] bf16
// ---------------------------------------------------------------------------
__global__ void k_attn(const int* __restrict__ bxi, const int* __restrict__ bei,
                       const float* __restrict__ X, const __bf16* __restrict__ hbf,
                       const float* __restrict__ a_src, const float* __restrict__ a_dst,
                       const float* __restrict__ gbias, __bf16* __restrict__ sig) {
  __shared__ float hF[NAREA * HCPAD];
  __shared__ float ssrc[NAREA][HH];
  __shared__ float sdst[HH];
  __shared__ float sc[NAREA][HH];
  __shared__ float alpha[NAREA][HH];
  __shared__ int   adj0[NAREA];
  __shared__ float out0[DD];
  int g = blockIdx.x, tid = threadIdx.x;
  const __bf16* hb = hbf + (long)g * NAREA * HCPAD;
  for (int e = tid; e < NAREA * HCPAD; e += 128) hF[e] = (float)hb[e];
  if (tid < NAREA) adj0[tid] = (tid == 0) ? 1 : 0;
  __syncthreads();
  {
    int j = tid >> 2, hh = tid & 3;
    float s = 0.f;
    for (int c = 0; c < CC; ++c) s += hF[j * HCPAD + hh * CC + c] * a_src[hh * CC + c];
    ssrc[j][hh] = s;
  }
  if (tid < HH) {
    float s = 0.f;
    for (int c = 0; c < CC; ++c) s += hF[tid * CC + c] * a_dst[tid * CC + c];
    sdst[tid] = s;
  }
  const int* eb = bei + (long)g * 2 * EE;
  for (int e = tid; e < EE; e += 128) {
    int srcn = eb[e], dstn = eb[EE + e];
    if (dstn == 0) adj0[srcn] = 1;    // benign race: only writes 1
  }
  __syncthreads();
  {
    int j = tid >> 2, hh = tid & 3;
    float s = sdst[hh] + ssrc[j][hh];
    s = (s > 0.f) ? s : 0.2f * s;     // leaky_relu(0.2) before mask
    if (!adj0[j]) s = -1e9f;
    sc[j][hh] = s;
  }
  __syncthreads();
  if (tid < HH) {
    float mx = -1e30f;
    for (int j = 0; j < NAREA; ++j) mx = fmaxf(mx, sc[j][tid]);
    float sum = 0.f;
    for (int j = 0; j < NAREA; ++j) { float e2 = __expf(sc[j][tid] - mx); alpha[j][tid] = e2; sum += e2; }
    float inv = 1.f / sum;
    for (int j = 0; j < NAREA; ++j) alpha[j][tid] *= inv;
  }
  __syncthreads();
  for (int hc = tid; hc < DD; hc += 128) {
    int hh = hc / CC;
    float s = 0.f;
    for (int j = 0; j < NAREA; ++j) s += alpha[j][hh] * hF[j * HCPAD + hc];
    out0[hc] = s + gbias[hc];
  }
  __syncthreads();
  int b = g / TT, t = g % TT;
  __bf16* dst = sig + ((long)t * BB + b) * SIGPAD;
  int idx0 = bxi[g * NAREA + 0];
  for (int i = tid; i < SIGPAD; i += 128) {
    float v;
    if (i < DD)         v = X[(long)idx0 * DD + i];
    else if (i < 2*DD)  v = out0[i - DD];
    else                v = 0.f;
    dst[i] = (__bf16)v;
  }
}

// ---------------------------------------------------------------------------
// LSTM state init
// ---------------------------------------------------------------------------
__global__ void k_zeroState(__bf16* hA, __bf16* hB, float* c, int n) {
  int i = blockIdx.x * blockDim.x + threadIdx.x;
  if (i < n) { hA[i] = (__bf16)0.f; hB[i] = (__bf16)0.f; c[i] = 0.f; }
}

// ---------------------------------------------------------------------------
// One LSTM timestep. Block = 32 units, all 4 gates (8 n-tiles x 2 m-tiles =
// 16 WMMA tiles over 8 waves; each wave's 2 tiles share one A fragment).
// h ping-pongs across step launches (no cross-block h hazard).
// ---------------------------------------------------------------------------
__global__ void k_lstmStep(const __bf16* __restrict__ xseq, int t, int Kx, int hs,
                           const __bf16* __restrict__ Wih, const __bf16* __restrict__ Whh,
                           const float* __restrict__ bih, const float* __restrict__ bhh,
                           const __bf16* __restrict__ hin, __bf16* __restrict__ hout,
                           float* __restrict__ cst,
                           __bf16* __restrict__ outseq, int outStride) {
  __shared__ float gates[BB * 128];               // [m][nt8*16 + n] == [m][gate*32+ul]
  int u0 = blockIdx.x * 32;
  int NT = hs >> 2;                                // 4hs/16 n-tiles total
  int tid = threadIdx.x, w = tid >> 5;
  const __bf16* xt = xseq + (long)t * BB * Kx;
  int mt = w & 1;
  int nt8a = w >> 1;                               // this wave's two n-tiles
  int nt8b = nt8a + 4;
  // global n-tile indices (gate gi = nt8>>1, half j = nt8&1)
  int ntgA = (nt8a >> 1) * (hs >> 4) + (u0 >> 4) + (nt8a & 1);
  int ntgB = (nt8b >> 1) * (hs >> 4) + (u0 >> 4) + (nt8b & 1);
  v8f accA = {}, accB = {};
  int kxT = Kx / 32, hsT = hs / 32;
  for (int kt = 0; kt < kxT; ++kt) {
    v16bf a = ldA(xt + mt * 16 * Kx + kt * 32, Kx);
    v16bf b0 = ldBp(Wih + (((long)kt * NT + ntgA) << 9));
    v16bf b1 = ldBp(Wih + (((long)kt * NT + ntgB) << 9));
    accA = wmma_bf(a, b0, accA);
    accB = wmma_bf(a, b1, accB);
  }
  for (int kt = 0; kt < hsT; ++kt) {
    v16bf a = ldA(hin + mt * 16 * hs + kt * 32, hs);
    v16bf b0 = ldBp(Whh + (((long)kt * NT + ntgA) << 9));
    v16bf b1 = ldBp(Whh + (((long)kt * NT + ntgB) << 9));
    accA = wmma_bf(a, b0, accA);
    accB = wmma_bf(a, b1, accB);
  }
  int lane = tid & 31, n = lane & 15, mh = (lane >> 4) * 8;
#pragma unroll
  for (int r = 0; r < 8; ++r) {
    int m = mt * 16 + mh + r;
    gates[m * 128 + nt8a * 16 + n] = accA[r];
    gates[m * 128 + nt8b * 16 + n] = accB[r];
  }
  __syncthreads();
  for (int p = tid; p < BB * 32; p += 256) {
    int m = p >> 5, ul = p & 31;
    int uu = u0 + ul;
    float gi_ = gates[m * 128 +   0 + ul] + bih[0 * hs + uu] + bhh[0 * hs + uu];
    float gf  = gates[m * 128 +  32 + ul] + bih[1 * hs + uu] + bhh[1 * hs + uu];
    float gg  = gates[m * 128 +  64 + ul] + bih[2 * hs + uu] + bhh[2 * hs + uu];
    float go  = gates[m * 128 +  96 + ul] + bih[3 * hs + uu] + bhh[3 * hs + uu];
    float si = 1.f / (1.f + __expf(-gi_));
    float sf = 1.f / (1.f + __expf(-gf));
    float so = 1.f / (1.f + __expf(-go));
    float tg = tanhf(gg);
    float cold = cst[m * hs + uu];
    float cnew = sf * cold + si * tg;
    float hnew = so * tanhf(cnew);
    cst[m * hs + uu] = cnew;
    hout[m * hs + uu] = (__bf16)hnew;
    outseq[((long)t * BB + m) * outStride + uu] = (__bf16)hnew;
  }
}

// ---------------------------------------------------------------------------
// Final GEMM: logits[2048 x 50000] = feats[2048 x 512] @ W2 + b
// Block = 32x256 tile; A staged in LDS; B streamed from packed tiles (L2).
// A wave's 4 tiles share one A fragment per k-tile.
// ---------------------------------------------------------------------------
__global__ void k_finalGemm(const __bf16* __restrict__ feats, const __bf16* __restrict__ W2p,
                            const float* __restrict__ bias, float* __restrict__ out) {
  __shared__ __align__(16) __bf16 Alds[32 * 512];  // 32 KB
  int m0 = blockIdx.x * 32;
  int nb16 = blockIdx.y * 16;                      // first n-tile of this block
  int tid = threadIdx.x, w = tid >> 5;
  for (int e = tid; e < 32 * 512 / 8; e += 256)
    ((v8bf*)Alds)[e] = ((const v8bf*)(feats + (long)m0 * 512))[e];
  __syncthreads();
  int mt = w & 1;
  int ntl = w >> 1;                                // wave's n-tiles: ntl + 4u
  v8f acc[4] = {};
  for (int kt = 0; kt < 16; ++kt) {
    v16bf a = ldA(Alds + mt * 16 * 512 + kt * 32, 512);
    if (kt < 15)
      __builtin_prefetch(W2p + (((long)(kt + 1) * NT2 + nb16) << 9), 0, 1);
#pragma unroll
    for (int u = 0; u < 4; ++u) {
      int ntg = nb16 + ntl + 4 * u;
      if (ntg < NT2) {                             // wave-uniform branch
        v16bf b = ldBp(W2p + (((long)kt * NT2 + ntg) << 9));
        acc[u] = wmma_bf(a, b, acc[u]);
      }
    }
  }
  int lane = tid & 31, n = lane & 15, mh = (lane >> 4) * 8;
#pragma unroll
  for (int u = 0; u < 4; ++u) {
    int ntg = nb16 + ntl + 4 * u;
    if (ntg < NT2) {
      int ng = ntg * 16 + n;
#pragma unroll
      for (int r = 0; r < 8; ++r) {
        int m = m0 + mt * 16 + mh + r;
        out[(long)m * NGLOB + ng] = acc[u][r] + bias[ng];
      }
    }
  }
}

// ---------------------------------------------------------------------------
// Row-wise log-softmax over 50000, one block per row
// ---------------------------------------------------------------------------
__global__ void k_logsoftmax(float* __restrict__ out) {
  __shared__ float red[256];
  int r = blockIdx.x, tid = threadIdx.x;
  float* row = out + (long)r * NGLOB;
  float mx = -1e30f;
  for (int i = tid; i < NGLOB; i += 256) mx = fmaxf(mx, row[i]);
  red[tid] = mx; __syncthreads();
  for (int s = 128; s > 0; s >>= 1) { if (tid < s) red[tid] = fmaxf(red[tid], red[tid + s]); __syncthreads(); }
  mx = red[0]; __syncthreads();
  float sum = 0.f;
  for (int i = tid; i < NGLOB; i += 256) sum += __expf(row[i] - mx);
  red[tid] = sum; __syncthreads();
  for (int s = 128; s > 0; s >>= 1) { if (tid < s) red[tid] += red[tid + s]; __syncthreads(); }
  float lse = mx + logf(red[0]);
  __syncthreads();
  for (int i = tid; i < NGLOB; i += 256) row[i] -= lse;
}

// ---------------------------------------------------------------------------
extern "C" void kernel_launch(void* const* d_in, const int* in_sizes, int n_in,
                              void* d_out, int out_size, void* d_ws, size_t ws_size,
                              hipStream_t stream) {
  (void)in_sizes; (void)n_in; (void)out_size; (void)ws_size;
  const int*   bxi  = (const int*)d_in[0];
  const int*   bei  = (const int*)d_in[1];
  const float* X    = (const float*)d_in[2];
  const float* gW   = (const float*)d_in[3];
  const float* aS   = (const float*)d_in[4];
  const float* aD   = (const float*)d_in[5];
  const float* gB   = (const float*)d_in[6];
  const float* Wih0 = (const float*)d_in[7];
  const float* Whh0 = (const float*)d_in[8];
  const float* bih0 = (const float*)d_in[9];
  const float* bhh0 = (const float*)d_in[10];
  const float* Wih1 = (const float*)d_in[11];
  const float* Whh1 = (const float*)d_in[12];
  const float* bih1 = (const float*)d_in[13];
  const float* bhh1 = (const float*)d_in[14];
  const float* Wih2 = (const float*)d_in[15];
  const float* Whh2 = (const float*)d_in[16];
  const float* bih2 = (const float*)d_in[17];
  const float* bhh2 = (const float*)d_in[18];
  const float* W2   = (const float*)d_in[19];
  const float* b2   = (const float*)d_in[20];
  float* out = (float*)d_out;

  // workspace layout (~190 MB total)
  char* ws = (char*)d_ws;
  size_t o = 0;
  auto alloc = [&](size_t sz) { size_t r = o; o += (sz + 255) & ~(size_t)255; return r; };
  __bf16* Xbf  = (__bf16*)(ws + alloc(75000UL * DPAD * 2));
  __bf16* Wgp  = (__bf16*)(ws + alloc(10UL * 19 * 512 * 2));
  __bf16* hbf  = (__bf16*)(ws + alloc((size_t)GG * NAREA * HCPAD * 2));
  __bf16* sig  = (__bf16*)(ws + alloc((size_t)TT * BB * SIGPAD * 2));
  __bf16* Wib0 = (__bf16*)(ws + alloc(19UL * 256 * 512 * 2));   // 608 x 4096
  __bf16* Whb0 = (__bf16*)(ws + alloc(32UL * 256 * 512 * 2));   // 1024 x 4096
  __bf16* Wib1 = (__bf16*)(ws + alloc(32UL * 256 * 512 * 2));
  __bf16* Whb1 = (__bf16*)(ws + alloc(32UL * 256 * 512 * 2));
  __bf16* Wib2 = (__bf16*)(ws + alloc(32UL * 128 * 512 * 2));   // 1024 x 2048
  __bf16* Whb2 = (__bf16*)(ws + alloc(16UL * 128 * 512 * 2));   //  512 x 2048
  __bf16* x1    = (__bf16*)(ws + alloc((size_t)TT * BB * 1024 * 2));
  __bf16* x2    = (__bf16*)(ws + alloc((size_t)TT * BB * 1024 * 2));
  __bf16* feats = (__bf16*)(ws + alloc((size_t)TT * BB * 512 * 2));
  __bf16* hA   = (__bf16*)(ws + alloc(32UL * 1024 * 2));
  __bf16* hB   = (__bf16*)(ws + alloc(32UL * 1024 * 2));
  float*  cst  = (float*)(ws + alloc(32UL * 1024 * 4));
  __bf16* W2p  = (__bf16*)(ws + alloc(16UL * NT2 * 512 * 2));   // 512 x 50000

  // ---- weight packing ----
  k_cvtX<<<75000, DPAD, 0, stream>>>(X, Xbf);
  auto packFrag = [&](const float* src, __bf16* dst, int K, int N,
                      int KT, int NT, int ldW, int trans) {
    long tot = (long)KT * NT * 512;
    k_packFrag<<<(unsigned)((tot + 255) / 256), 256, 0, stream>>>(
        src, dst, K, N, KT, NT, ldW, trans);
  };
  packFrag(gW,   Wgp,  300,  300,   10, 19,   300,  0);  // lin_w is [K][N]
  packFrag(Wih0, Wib0, 600,  4096,  19, 256,  600,  1);  // torch [N][K]
  packFrag(Whh0, Whb0, 1024, 4096,  32, 256,  1024, 1);
  packFrag(Wih1, Wib1, 1024, 4096,  32, 256,  1024, 1);
  packFrag(Whh1, Whb1, 1024, 4096,  32, 256,  1024, 1);
  packFrag(Wih2, Wib2, 1024, 2048,  32, 128,  1024, 1);
  packFrag(Whh2, Whb2, 512,  2048,  16, 128,  512,  1);
  packFrag(W2,   W2p,  512,  NGLOB, 16, NT2,  512,  1);

  // ---- GAT ----
  k_gatGemm<<<GG, 256, 0, stream>>>(bxi, Xbf, Wgp, hbf);
  k_attn<<<GG, 128, 0, stream>>>(bxi, bei, X, hbf, aS, aD, gB, sig);

  // ---- LSTM (3 layers, 64 dependent steps each, ping-pong h) ----
  struct Layer {
    const __bf16* xs; int Kx, hs;
    const __bf16 *Wib, *Whb; const float *bi, *bh;
    __bf16* outp; int ostr;
  };
  Layer L[3] = {
    { sig, SIGPAD, 1024, Wib0, Whb0, bih0, bhh0, x1,    1024 },
    { x1,  1024,   1024, Wib1, Whb1, bih1, bhh1, x2,    1024 },
    { x2,  1024,   512,  Wib2, Whb2, bih2, bhh2, feats, 512  },
  };
  for (int l = 0; l < 3; ++l) {
    int n = 32 * L[l].hs;
    k_zeroState<<<(n + 255) / 256, 256, 0, stream>>>(hA, hB, cst, n);
    for (int t = 0; t < TT; ++t) {
      const __bf16* hin = (t & 1) ? hB : hA;
      __bf16*      hout = (t & 1) ? hA : hB;
      k_lstmStep<<<L[l].hs / 32, 256, 0, stream>>>(
          L[l].xs, t, L[l].Kx, L[l].hs, L[l].Wib, L[l].Whb, L[l].bi, L[l].bh,
          hin, hout, cst, L[l].outp, L[l].ostr);
    }
  }

  // ---- classifier + log_softmax ----
  dim3 fg(GG / 32, (NT2 + 15) / 16);
  k_finalGemm<<<fg, 256, 0, stream>>>(feats, W2p, b2, out);
  k_logsoftmax<<<GG, 256, 0, stream>>>(out);
}
